// RAMMultiHeadKV_27668179321268
// MI455X (gfx1250) — matrix operand order repaired and checked by reading, not compiled, not run on hardware
//
#include <hip/hip_runtime.h>

typedef __attribute__((ext_vector_type(16))) _Float16 v16h;
typedef __attribute__((ext_vector_type(8)))  float    v8f;

#define NUM_HEADS  256
#define V_BITS     16
#define NEURONS    12
#define INPUT_BITS 32
#define N_WINDOWS  2048
#define NTHREADS   1024
#define NWAVES     (NTHREADS / 32)
#define NTILES     (N_WINDOWS / 16)

__global__ __launch_bounds__(NTHREADS)
void ram_mhkv_kernel(const int* __restrict__ input_bits,
                     const int* __restrict__ conn_state,
                     const int* __restrict__ mem_state,
                     const int* __restrict__ conn_out,
                     const float* __restrict__ mem_out,
                     float* __restrict__ out)
{
    __shared__ unsigned       s_key[N_WINDOWS];      // packed 16-bit key per window
    __shared__ unsigned       s_wr [N_WINDOWS];      // count of nonzero value bits
    __shared__ unsigned       s_ms [NEURONS * 128];  // mem_state[h] bit-packed (12 x 4096 bits)
    __shared__ int            s_cs [NEURONS * NEURONS];
    __shared__ unsigned short s_mlist[N_WINDOWS];    // ordered write windows hitting head h
    __shared__ int            s_scan[256];
    __shared__ int            s_tq, s_tqr, s_head, s_nmatch;
    __shared__ unsigned       s_nm[2];               // double-buffered new-state mask

    const int tid  = (int)threadIdx.x;
    const int lane = tid & 31;
    const int wave = tid >> 5;

    if (tid == 0) { s_tq = -1; s_nm[0] = 0u; s_nm[1] = 0u; }

    // ---------------- Phase A: WMMA bit-packing front-end ----------------
    // D = A x B; A = window bits (16x32, f16 0/1), B col0 = 2^(15-k) (k<16),
    // B col1 = 1.0 for k>=16 (write-bit count). f32 accumulate => exact ints.
    v16h b;
    for (int i = 0; i < 16; ++i) b[i] = (_Float16)0.0f;
    if (lane == 0)  { for (int k = 0; k < 16; ++k) b[k] = (_Float16)(float)(1u << (15 - k)); }
    if (lane == 17) { for (int k = 0; k < 16; ++k) b[k] = (_Float16)1.0f; }

    for (int t = wave; t < NTILES; t += NWAVES) {
        const int row = lane & 15;
        const int hi  = (lane >> 4) & 1;          // lane half selects K groups
        const int w   = t * 16 + row;
        const int* wb = input_bits + w * INPUT_BITS;
        const int k1  = hi * 16;                  // K = 0..7  or 16..23
        const int k2  = hi * 16 + 8;              // K = 8..15 or 24..31
        v16h a;
        for (int j = 0; j < 8; ++j) a[j]     = (wb[k1 + j] != 0) ? (_Float16)1.0f : (_Float16)0.0f;
        for (int j = 0; j < 8; ++j) a[8 + j] = (wb[k2 + j] != 0) ? (_Float16)1.0f : (_Float16)0.0f;
        v8f c = {};
        c = __builtin_amdgcn_wmma_f32_16x16x32_f16(false, a, false, b, (short)0, c, false, false);
        const int m0 = t * 16 + hi * 8;           // C layout: VGPR r -> M = r + 8*hi
        if ((lane & 15) == 0) { for (int r = 0; r < 8; ++r) s_key[m0 + r] = (unsigned)(c[r] + 0.25f); }
        if ((lane & 15) == 1) { for (int r = 0; r < 8; ++r) s_wr [m0 + r] = (unsigned)(c[r] + 0.25f); }
    }
    __syncthreads();

    // ---------------- Phase B: last query window + its head ----------------
    for (int w = tid; w < N_WINDOWS; w += NTHREADS)
        if (s_wr[w] == 0) atomicMax(&s_tq, w);
    __syncthreads();
    if (tid == 0) {
        const int tq = (s_tq < 0) ? (N_WINDOWS - 1) : s_tq;   // no query -> last_out path
        s_tqr  = tq;
        s_head = (int)(s_key[tq] & (NUM_HEADS - 1));
    }
    __syncthreads();
    const int h  = s_head;
    const int tq = s_tqr;

    // ---------------- Phase C: stage head-h tables into LDS ----------------
    if (tid < NEURONS * NEURONS) s_cs[tid] = conn_state[h * NEURONS * NEURONS + tid];
    {
        const int base = h * (NEURONS * 4096);
        for (int idx = tid; idx < NEURONS * 4096; idx += NTHREADS) {   // 48 exact iters
            const int v = mem_state[base + idx];
            const unsigned mask = (unsigned)__builtin_amdgcn_ballot_w32(v != 0);
            if (lane == 0) s_ms[idx >> 5] = mask;
        }
    }
    __syncthreads();

    // ---------------- Phase D: ordered compaction of matching writes ----------------
    if (tid < 256) {
        int cnt = 0;
        const int w0 = tid * 8;
        for (int k = 0; k < 8; ++k) {
            const int w = w0 + k;
            if (w < tq && s_wr[w] != 0 && (s_key[w] & (NUM_HEADS - 1)) == (unsigned)h) ++cnt;
        }
        s_scan[tid] = cnt;
    }
    __syncthreads();
    if (tid == 0) {
        int tot = 0;
        for (int i = 0; i < 256; ++i) { const int c = s_scan[i]; s_scan[i] = tot; tot += c; }
        s_nmatch = tot;
    }
    __syncthreads();
    if (tid < 256) {
        int off = s_scan[tid];
        const int w0 = tid * 8;
        for (int k = 0; k < 8; ++k) {
            const int w = w0 + k;
            if (w < tq && s_wr[w] != 0 && (s_key[w] & (NUM_HEADS - 1)) == (unsigned)h)
                s_mlist[off++] = (unsigned short)w;
        }
    }
    __syncthreads();

    // ---------------- Phase E: sequential replay (LDS-only) ----------------
    const int nm = s_nmatch;
    unsigned st = 0u;                         // 12-bit state of head h
    for (int s = 0; s <= nm; ++s) {           // s == nm is the query/terminal step itself
        const int w        = (s < nm) ? (int)s_mlist[s] : tq;
        const unsigned key = s_key[w];
        const int p        = s & 1;
        if (tid < NEURONS) {
            unsigned addr = 0u;
            for (int j = 0; j < NEURONS; ++j) {
                const int cix = s_cs[tid * NEURONS + j];
                const unsigned bb = (cix < 16) ? ((key >> (15 - cix)) & 1u)
                                               : ((st >> (cix - 16)) & 1u);
                addr = (addr << 1) | bb;
            }
            const unsigned bit = (s_ms[tid * 128 + (addr >> 5)] >> (addr & 31)) & 1u;
            atomicOr(&s_nm[p], bit << tid);
        }
        if (tid == 0) s_nm[1 - p] = 0u;       // prepare next buffer (disjoint address)
        __syncthreads();
        st = s_nm[p];                          // new_st of this step (all threads)
        __syncthreads();                       // protect read vs. next-iter reset
    }

    // ---------------- Phase F: 16 outputs ----------------
    if (tid < V_BITS) {
        unsigned oaddr = 0u;
        const int* co = conn_out + (h * V_BITS + tid) * NEURONS;
        for (int j = 0; j < NEURONS; ++j) oaddr = (oaddr << 1) | ((st >> co[j]) & 1u);
        out[tid] = mem_out[(h * V_BITS + tid) * 4096 + oaddr];
    }
}

extern "C" void kernel_launch(void* const* d_in, const int* in_sizes, int n_in,
                              void* d_out, int out_size, void* d_ws, size_t ws_size,
                              hipStream_t stream) {
    (void)in_sizes; (void)n_in; (void)d_ws; (void)ws_size; (void)out_size;
    const int*   input_bits = (const int*)  d_in[0];
    const int*   conn_state = (const int*)  d_in[1];
    const int*   mem_state  = (const int*)  d_in[2];
    const int*   conn_out   = (const int*)  d_in[3];
    const float* mem_out    = (const float*)d_in[4];
    float*       out        = (float*)d_out;

    ram_mhkv_kernel<<<1, NTHREADS, 0, stream>>>(input_bits, conn_state, mem_state,
                                                conn_out, mem_out, out);
}
